// QuantizedAttention_14388140441831
// MI455X (gfx1250) — matrix operand order repaired and checked by reading, not compiled, not run on hardware
//
#include <hip/hip_runtime.h>
#include <cstdint>
#include <cstddef>

typedef unsigned short u16;
typedef float f32x4 __attribute__((ext_vector_type(4)));
typedef float v8f   __attribute__((ext_vector_type(8)));
typedef __bf16 v16bf __attribute__((ext_vector_type(16)));

union Frag { v16bf v; f32x4 f[2]; };

#define CB 2
#define CS 2048
#define CD 1024
#define CH 16
#define CDh 64
#define CM (CB*CS)          // 4096 rows
#define CN 1024             // H*Dh == D
#define IGNORE_VAL (-1000.0f)

static __device__ __forceinline__ u16 f32_to_bf16(float x) {
    unsigned u = __float_as_uint(x);
    u += 0x7FFFu + ((u >> 16) & 1u);   // round-to-nearest-even
    return (u16)(u >> 16);
}

static __device__ __forceinline__ float qdq_e4m3(float w, float amax) {
    float scale = 448.0f / fmaxf(amax, 1e-12f);
    float v = w * scale;
    float a = fabsf(v);
    float e = floorf(__log2f(fmaxf(a, 1e-30f)));
    e = fminf(fmaxf(e, -6.0f), 8.0f);
    float step = exp2f(e - 3.0f);
    float q = rintf(v / step) * step;
    q = fminf(fmaxf(q, -448.0f), 448.0f);
    return q / scale;
}

// ---------------- amax reduction ----------------
__global__ __launch_bounds__(256) void reduce_absmax_kernel(
    const float* __restrict__ x, int n, float* __restrict__ out) {
    __shared__ float red[256];
    float m = 0.f;
    for (int i = blockIdx.x * 256 + threadIdx.x; i < n; i += gridDim.x * 256)
        m = fmaxf(m, fabsf(x[i]));
    red[threadIdx.x] = m;
    __syncthreads();
    for (int s = 128; s > 0; s >>= 1) {
        if ((int)threadIdx.x < s) red[threadIdx.x] = fmaxf(red[threadIdx.x], red[threadIdx.x + s]);
        __syncthreads();
    }
    if (threadIdx.x == 0) atomicMax((unsigned int*)out, __float_as_uint(red[0]));
}

// --------- dequant W (H,D,Dh) -> bf16 Wt[(h*Dh+e)][d] ----------
__global__ __launch_bounds__(256) void dequant_weight_kernel(
    const float* __restrict__ W, const float* __restrict__ amax,
    u16* __restrict__ Wt) {
    int idx = blockIdx.x * 256 + threadIdx.x;          // over H*D*Dh = 1M
    if (idx >= CH * CD * CDh) return;
    float am = *amax;
    int h = idx >> 16;            // / (CD*CDh)
    int rem = idx & 65535;
    int d = rem >> 6;
    int e = rem & 63;
    Wt[(size_t)((h << 6) + e) * CD + d] = f32_to_bf16(qdq_e4m3(W[idx], am));
}

__global__ __launch_bounds__(256) void dequant_bias_kernel(
    const float* __restrict__ b, const float* __restrict__ amax,
    float* __restrict__ bd) {
    int i = blockIdx.x * 256 + threadIdx.x;
    if (i < CH * CDh) bd[i] = qdq_e4m3(b[i], *amax);
}

__global__ __launch_bounds__(256) void convert_bf16_kernel(
    const float* __restrict__ x, u16* __restrict__ y, int n) {
    int i = blockIdx.x * 256 + threadIdx.x;
    if (i < n) y[i] = f32_to_bf16(x[i]);
}

// W_O (H,Dh,D) -> bf16 WoT[d][(h*Dh+e)]
__global__ __launch_bounds__(256) void convert_wo_kernel(
    const float* __restrict__ Wo, u16* __restrict__ WoT) {
    int idx = blockIdx.x * 256 + threadIdx.x;          // over 1M, idx = he*1024 + d
    if (idx >= CH * CDh * CD) return;
    int he = idx >> 10;
    int d = idx & 1023;
    WoT[(size_t)d * CN + he] = f32_to_bf16(Wo[idx]);
}

// ---------------- projection GEMM + bias + rotary ----------------
// X[M][K=1024] bf16, Wt[N=1024][K=1024] bf16 (K-major), out dst[B,H,S,Dh] bf16
// Block tile 128(M) x 64(N), 4 waves, each wave 32x64 (2x4 WMMA tiles), k-step 32.
__global__ __launch_bounds__(128) void proj_gemm_kernel(
    const u16* __restrict__ Xb, const u16* __restrict__ Wt,
    const float* __restrict__ biasd, u16* __restrict__ dst, int apply_rope) {
    __shared__ alignas(16) u16 As[128 * 32];
    __shared__ alignas(16) u16 Bs[64 * 32];
    const int tid = threadIdx.x;
    const int lane = tid & 31, w = tid >> 5;
    const int laneLo = lane & 15;
    const int hi = (lane >> 4) & 1;
    const int n0 = blockIdx.x * 64;    // head h = blockIdx.x
    const int m0 = blockIdx.y * 128;

    v8f acc[2][4];
    const v8f vzero = {0.f, 0.f, 0.f, 0.f, 0.f, 0.f, 0.f, 0.f};
#pragma unroll
    for (int u = 0; u < 2; ++u)
#pragma unroll
        for (int t = 0; t < 4; ++t) acc[u][t] = vzero;

    const int brow = tid >> 1, bhalf = tid & 1;   // B staging: 64 rows x 2 halves
    for (int k0 = 0; k0 < CD; k0 += 32) {
        __syncthreads();
        {
            // A: 128 rows x 32 k, one 64B row per thread
            const f32x4* sa = (const f32x4*)(Xb + (size_t)(m0 + tid) * CD + k0);
            f32x4* da = (f32x4*)(As + tid * 32);
            da[0] = sa[0]; da[1] = sa[1]; da[2] = sa[2]; da[3] = sa[3];
            // B: 64 rows x 32 k, 32B per thread
            const f32x4* sb = (const f32x4*)(Wt + (size_t)(n0 + brow) * CD + k0 + bhalf * 16);
            f32x4* db = (f32x4*)(Bs + brow * 32 + bhalf * 16);
            db[0] = sb[0]; db[1] = sb[1];
            if (k0 + 32 < CD) {
                __builtin_prefetch(Xb + (size_t)(m0 + tid) * CD + k0 + 32, 0, 1);
                __builtin_prefetch(Wt + (size_t)(n0 + brow) * CD + k0 + 32, 0, 1);
            }
        }
        __syncthreads();
        Frag a[2];
#pragma unroll
        for (int u = 0; u < 2; ++u) {
            const int arow = w * 32 + u * 16 + laneLo;
            a[u].f[0] = *(const f32x4*)(As + arow * 32 + hi * 8);
            a[u].f[1] = *(const f32x4*)(As + arow * 32 + 16 + hi * 8);
        }
#pragma unroll
        for (int t = 0; t < 4; ++t) {
            Frag b;
            const int bn = t * 16 + laneLo;
            b.f[0] = *(const f32x4*)(Bs + bn * 32 + hi * 8);
            b.f[1] = *(const f32x4*)(Bs + bn * 32 + 16 + hi * 8);
#pragma unroll
            for (int u = 0; u < 2; ++u)
                acc[u][t] = __builtin_amdgcn_wmma_f32_16x16x32_bf16(
                    false, a[u].v, false, b.v, (short)0, acc[u][t], false, false);
        }
    }

    // epilogue: bias, rotary, store
    const int h = blockIdx.x;
#pragma unroll
    for (int u = 0; u < 2; ++u) {
        float outv[4][8];
#pragma unroll
        for (int t = 0; t < 4; ++t) {
            float bv = biasd[n0 + t * 16 + laneLo];
#pragma unroll
            for (int r = 0; r < 8; ++r) acc[u][t][r] += bv;
        }
        if (apply_rope) {
#pragma unroll
            for (int t = 0; t < 4; ++t) {
                const int e = t * 16 + laneLo;
                const int half_idx = e & 31;
                const float invf = __powf(10000.0f, -(float)half_idx * (1.0f / 32.0f));
#pragma unroll
                for (int r = 0; r < 8; ++r) {
                    const int m = m0 + w * 32 + u * 16 + r + hi * 8;
                    const int s = m & (CS - 1);
                    float sn, cs;
                    __sincosf((float)s * invf, &sn, &cs);
                    const float flip = (t < 2) ? -acc[u][t + 2][r] : acc[u][t - 2][r];
                    outv[t][r] = acc[u][t][r] * cs + flip * sn;
                }
            }
        } else {
#pragma unroll
            for (int t = 0; t < 4; ++t)
#pragma unroll
                for (int r = 0; r < 8; ++r) outv[t][r] = acc[u][t][r];
        }
#pragma unroll
        for (int t = 0; t < 4; ++t) {
            const int e = t * 16 + laneLo;
#pragma unroll
            for (int r = 0; r < 8; ++r) {
                const int m = m0 + w * 32 + u * 16 + r + hi * 8;
                const int b = m >> 11;
                const int s = m & (CS - 1);
                dst[(((size_t)b * CH + h) * CS + s) * CDh + e] = f32_to_bf16(outv[t][r]);
            }
        }
    }
}

// ---------------- flash attention (causal, IGNORE=-1000) ----------------
// q/k/v: [B,H,S,Dh] bf16 ; z out: [B,S,H,Dh] bf16
__global__ __launch_bounds__(128) void attn_kernel(
    const u16* __restrict__ qg, const u16* __restrict__ kg,
    const u16* __restrict__ vg, u16* __restrict__ z) {
    __shared__ alignas(16) u16 Ks[64 * 64];
    __shared__ alignas(16) u16 Vs[64 * 64];
    __shared__ alignas(16) u16 Pb[4 * 16 * 64];
    const int tid = threadIdx.x;
    const int lane = tid & 31, w = tid >> 5;
    const int laneLo = lane & 15;
    const int hi = (lane >> 4) & 1;
    const int qblk = blockIdx.x, h = blockIdx.y, b = blockIdx.z;
    const int qbase = qblk * 64;
    const size_t headoff = ((size_t)b * CH + h) * CS * CDh;
    const u16* qp = qg + headoff;
    const u16* kp = kg + headoff;
    const u16* vp = vg + headoff;

    // Q fragments loaded directly from global (row-major [S][64] matches A layout)
    Frag qf[2];
    const int qrow = qbase + w * 16 + laneLo;
#pragma unroll
    for (int i = 0; i < 2; ++i) {
        const int es = i * 32;
        qf[i].f[0] = *(const f32x4*)(qp + (size_t)qrow * CDh + es + hi * 8);
        qf[i].f[1] = *(const f32x4*)(qp + (size_t)qrow * CDh + es + 16 + hi * 8);
    }

    v8f acc[4];
    const v8f vzero = {0.f, 0.f, 0.f, 0.f, 0.f, 0.f, 0.f, 0.f};
#pragma unroll
    for (int t = 0; t < 4; ++t) acc[t] = vzero;
    float mi[8], li[8];
#pragma unroll
    for (int r = 0; r < 8; ++r) { mi[r] = -3.0e38f; li[r] = 0.f; }
    u16* Pw = Pb + w * 16 * 64;

    for (int kb = 0; kb <= qblk; ++kb) {
        __syncthreads();
        {   // stage K rows (row-major)
            const int row = tid >> 1, half = tid & 1;
            const f32x4* sk = (const f32x4*)(kp + (size_t)(kb * 64 + row) * CDh + half * 32);
            f32x4* dk = (f32x4*)(Ks + row * 64 + half * 32);
            dk[0] = sk[0]; dk[1] = sk[1]; dk[2] = sk[2]; dk[3] = sk[3];
            // stage V transposed: Vs[e][kcol]
            const int kcol = tid & 63;
            const int e0 = (tid >> 6) * 32;
            union { f32x4 f[4]; u16 u[32]; } tu;
            const f32x4* sv = (const f32x4*)(vp + (size_t)(kb * 64 + kcol) * CDh + e0);
            tu.f[0] = sv[0]; tu.f[1] = sv[1]; tu.f[2] = sv[2]; tu.f[3] = sv[3];
#pragma unroll
            for (int i = 0; i < 32; ++i) Vs[(e0 + i) * 64 + kcol] = tu.u[i];
            if (kb + 1 <= qblk)
                __builtin_prefetch(kp + (size_t)((kb + 1) * 64 + row) * CDh, 0, 1);
        }
        __syncthreads();

        // scores: 4 tiles of 16x16
        float sc[4][8];
#pragma unroll
        for (int t = 0; t < 4; ++t) {
            v8f s = vzero;
#pragma unroll
            for (int i = 0; i < 2; ++i) {
                Frag kf;
                const int kc = t * 16 + laneLo;
                kf.f[0] = *(const f32x4*)(Ks + kc * 64 + i * 32 + hi * 8);
                kf.f[1] = *(const f32x4*)(Ks + kc * 64 + i * 32 + 16 + hi * 8);
                s = __builtin_amdgcn_wmma_f32_16x16x32_bf16(
                    false, qf[i].v, false, kf.v, (short)0, s, false, false);
            }
#pragma unroll
            for (int r = 0; r < 8; ++r) {
                float v = s[r] * 0.125f;   // 1/sqrt(Dh)
                if (kb == qblk) {
                    const int kcg = kb * 64 + t * 16 + laneLo;
                    const int qrg = qbase + w * 16 + r + hi * 8;
                    if (kcg > qrg) v = IGNORE_VAL;
                }
                sc[t][r] = v;
            }
        }

        // online softmax per row
#pragma unroll
        for (int r = 0; r < 8; ++r) {
            float tm = fmaxf(fmaxf(sc[0][r], sc[1][r]), fmaxf(sc[2][r], sc[3][r]));
            tm = fmaxf(tm, __shfl_xor(tm, 1, 32));
            tm = fmaxf(tm, __shfl_xor(tm, 2, 32));
            tm = fmaxf(tm, __shfl_xor(tm, 4, 32));
            tm = fmaxf(tm, __shfl_xor(tm, 8, 32));
            const float mnew = fmaxf(mi[r], tm);
            const float corr = __expf(mi[r] - mnew);
            float ps = 0.f;
#pragma unroll
            for (int t = 0; t < 4; ++t) {
                const float p = __expf(sc[t][r] - mnew);
                sc[t][r] = p;
                ps += p;
            }
            ps += __shfl_xor(ps, 1, 32);
            ps += __shfl_xor(ps, 2, 32);
            ps += __shfl_xor(ps, 4, 32);
            ps += __shfl_xor(ps, 8, 32);
            li[r] = li[r] * corr + ps;
            mi[r] = mnew;
#pragma unroll
            for (int t = 0; t < 4; ++t) acc[t][r] *= corr;
            const int prow = r + hi * 8;
#pragma unroll
            for (int t = 0; t < 4; ++t)
                Pw[prow * 64 + t * 16 + laneLo] = f32_to_bf16(sc[t][r]);
        }
        asm volatile("s_wait_dscnt 0" ::: "memory");   // LDS write->read (same wave)

        // PV: acc += P (16x64) x V (64x64)
#pragma unroll
        for (int i = 0; i < 2; ++i) {
            Frag pf;
            pf.f[0] = *(const f32x4*)(Pw + laneLo * 64 + i * 32 + hi * 8);
            pf.f[1] = *(const f32x4*)(Pw + laneLo * 64 + i * 32 + 16 + hi * 8);
#pragma unroll
            for (int t = 0; t < 4; ++t) {
                Frag vf;
                const int e = t * 16 + laneLo;
                vf.f[0] = *(const f32x4*)(Vs + e * 64 + i * 32 + hi * 8);
                vf.f[1] = *(const f32x4*)(Vs + e * 64 + i * 32 + 16 + hi * 8);
                acc[t] = __builtin_amdgcn_wmma_f32_16x16x32_bf16(
                    false, pf.v, false, vf.v, (short)0, acc[t], false, false);
            }
        }
    }

    // normalize and store z[B,S,H,Dh]
#pragma unroll
    for (int t = 0; t < 4; ++t) {
#pragma unroll
        for (int r = 0; r < 8; ++r) {
            const int qrg = qbase + w * 16 + r + hi * 8;
            const float o = acc[t][r] / li[r];
            z[(((size_t)b * CS + qrg) * CH + h) * CDh + t * 16 + laneLo] = f32_to_bf16(o);
        }
    }
}

// ---------------- output projection GEMM (f32 out) ----------------
// Z[M][1024] bf16, WoT[N=1024][K=1024] bf16, out[M][1024] f32 (+b_O)
// Block tile 128(M) x 64(N), 4 waves, each wave 32x64, k-step 32.
__global__ __launch_bounds__(128) void out_gemm_kernel(
    const u16* __restrict__ Zb, const u16* __restrict__ WoT,
    const float* __restrict__ bO, float* __restrict__ out) {
    __shared__ alignas(16) u16 As[128 * 32];
    __shared__ alignas(16) u16 Bs[64 * 32];
    const int tid = threadIdx.x;
    const int lane = tid & 31, w = tid >> 5;
    const int laneLo = lane & 15;
    const int hi = (lane >> 4) & 1;
    const int n0 = blockIdx.x * 64;
    const int m0 = blockIdx.y * 128;

    v8f acc[2][4];
    const v8f vzero = {0.f, 0.f, 0.f, 0.f, 0.f, 0.f, 0.f, 0.f};
#pragma unroll
    for (int u = 0; u < 2; ++u)
#pragma unroll
        for (int t = 0; t < 4; ++t) acc[u][t] = vzero;

    const int brow = tid >> 1, bhalf = tid & 1;
    for (int k0 = 0; k0 < CN; k0 += 32) {
        __syncthreads();
        {
            const f32x4* sa = (const f32x4*)(Zb + (size_t)(m0 + tid) * CN + k0);
            f32x4* da = (f32x4*)(As + tid * 32);
            da[0] = sa[0]; da[1] = sa[1]; da[2] = sa[2]; da[3] = sa[3];
            const f32x4* sb = (const f32x4*)(WoT + (size_t)(n0 + brow) * CN + k0 + bhalf * 16);
            f32x4* db = (f32x4*)(Bs + brow * 32 + bhalf * 16);
            db[0] = sb[0]; db[1] = sb[1];
            if (k0 + 32 < CN) {
                __builtin_prefetch(Zb + (size_t)(m0 + tid) * CN + k0 + 32, 0, 1);
                __builtin_prefetch(WoT + (size_t)(n0 + brow) * CN + k0 + 32, 0, 1);
            }
        }
        __syncthreads();
        Frag a[2];
#pragma unroll
        for (int u = 0; u < 2; ++u) {
            const int arow = w * 32 + u * 16 + laneLo;
            a[u].f[0] = *(const f32x4*)(As + arow * 32 + hi * 8);
            a[u].f[1] = *(const f32x4*)(As + arow * 32 + 16 + hi * 8);
        }
#pragma unroll
        for (int t = 0; t < 4; ++t) {
            Frag bfr;
            const int bn = t * 16 + laneLo;
            bfr.f[0] = *(const f32x4*)(Bs + bn * 32 + hi * 8);
            bfr.f[1] = *(const f32x4*)(Bs + bn * 32 + 16 + hi * 8);
#pragma unroll
            for (int u = 0; u < 2; ++u)
                acc[u][t] = __builtin_amdgcn_wmma_f32_16x16x32_bf16(
                    false, a[u].v, false, bfr.v, (short)0, acc[u][t], false, false);
        }
    }
#pragma unroll
    for (int u = 0; u < 2; ++u)
#pragma unroll
        for (int t = 0; t < 4; ++t) {
            const int col = n0 + t * 16 + laneLo;
            const float bv = bO[col];
#pragma unroll
            for (int r = 0; r < 8; ++r) {
                const int m = m0 + w * 32 + u * 16 + r + hi * 8;
                out[(size_t)m * CD + col] = acc[u][t][r] + bv;
            }
        }
}

// ---------------- host launcher ----------------
extern "C" void kernel_launch(void* const* d_in, const int* in_sizes, int n_in,
                              void* d_out, int out_size, void* d_ws, size_t ws_size,
                              hipStream_t stream) {
    (void)in_sizes; (void)n_in; (void)out_size; (void)ws_size;
    const float* Xq_f = (const float*)d_in[0];
    const float* Xk_f = (const float*)d_in[1];
    const float* Xv_f = (const float*)d_in[2];
    const float* WQ = (const float*)d_in[3];
    const float* WK = (const float*)d_in[4];
    const float* WV = (const float*)d_in[5];
    const float* WO = (const float*)d_in[6];
    const float* bQ = (const float*)d_in[7];
    const float* bK = (const float*)d_in[8];
    const float* bV = (const float*)d_in[9];
    const float* bO = (const float*)d_in[10];

    char* ws = (char*)d_ws;
    size_t off = 0;
    auto take = [&](size_t bytes) -> void* {
        void* p = ws + off;
        off = (off + bytes + 255) & ~(size_t)255;
        return p;
    };
    float* amax = (float*)take(6 * sizeof(float));   // [Wq,Wk,Wv,bq,bk,bv]
    float* bqd = (float*)take(1024 * sizeof(float));
    float* bkd = (float*)take(1024 * sizeof(float));
    float* bvd = (float*)take(1024 * sizeof(float));
    u16* WqT = (u16*)take((size_t)CN * CD * 2);
    u16* WkT = (u16*)take((size_t)CN * CD * 2);
    u16* WvT = (u16*)take((size_t)CN * CD * 2);
    u16* WoT = (u16*)take((size_t)CD * CN * 2);
    u16* Xqb = (u16*)take((size_t)CM * CD * 2);
    u16* Xkb = (u16*)take((size_t)CM * CD * 2);
    u16* Xvb = (u16*)take((size_t)CM * CD * 2);
    u16* qr_ = (u16*)take((size_t)CM * CN * 2);
    u16* kr_ = (u16*)take((size_t)CM * CN * 2);
    u16* vr_ = (u16*)take((size_t)CM * CN * 2);
    u16* Zb = (u16*)take((size_t)CM * CN * 2);

    hipMemsetAsync(amax, 0, 6 * sizeof(float), stream);

    const int nW = CH * CD * CDh;     // 1M
    const int nB = CH * CDh;          // 1024
    reduce_absmax_kernel<<<1024, 256, 0, stream>>>(WQ, nW, amax + 0);
    reduce_absmax_kernel<<<1024, 256, 0, stream>>>(WK, nW, amax + 1);
    reduce_absmax_kernel<<<1024, 256, 0, stream>>>(WV, nW, amax + 2);
    reduce_absmax_kernel<<<4, 256, 0, stream>>>(bQ, nB, amax + 3);
    reduce_absmax_kernel<<<4, 256, 0, stream>>>(bK, nB, amax + 4);
    reduce_absmax_kernel<<<4, 256, 0, stream>>>(bV, nB, amax + 5);

    dequant_weight_kernel<<<nW / 256, 256, 0, stream>>>(WQ, amax + 0, WqT);
    dequant_weight_kernel<<<nW / 256, 256, 0, stream>>>(WK, amax + 1, WkT);
    dequant_weight_kernel<<<nW / 256, 256, 0, stream>>>(WV, amax + 2, WvT);
    dequant_bias_kernel<<<4, 256, 0, stream>>>(bQ, amax + 3, bqd);
    dequant_bias_kernel<<<4, 256, 0, stream>>>(bK, amax + 4, bkd);
    dequant_bias_kernel<<<4, 256, 0, stream>>>(bV, amax + 5, bvd);

    const int nX = CM * CD;           // 4M
    convert_bf16_kernel<<<nX / 256, 256, 0, stream>>>(Xq_f, Xqb, nX);
    convert_bf16_kernel<<<nX / 256, 256, 0, stream>>>(Xk_f, Xkb, nX);
    convert_bf16_kernel<<<nX / 256, 256, 0, stream>>>(Xv_f, Xvb, nX);
    convert_wo_kernel<<<(CH * CDh * CD) / 256, 256, 0, stream>>>(WO, WoT);

    dim3 gGemm(CN / 64, CM / 128);    // (16, 32)
    proj_gemm_kernel<<<gGemm, 128, 0, stream>>>(Xqb, WqT, bqd, qr_, 1);
    proj_gemm_kernel<<<gGemm, 128, 0, stream>>>(Xkb, WkT, bkd, kr_, 1);
    proj_gemm_kernel<<<gGemm, 128, 0, stream>>>(Xvb, WvT, bvd, vr_, 0);

    dim3 gAttn(CS / 64, CH, CB);      // (32, 16, 2)
    attn_kernel<<<gAttn, 128, 0, stream>>>(qr_, kr_, vr_, Zb);

    out_gemm_kernel<<<gGemm, 128, 0, stream>>>(Zb, WoT, bO, (float*)d_out);
}
